// WAutoEncoder_63282048139984
// MI455X (gfx1250) — matrix operand order repaired and checked by reading, not compiled, not run on hardware
//
#include <hip/hip_runtime.h>
#include <hip/hip_bf16.h>

typedef __attribute__((ext_vector_type(16))) _Float16 v16h;
typedef __attribute__((ext_vector_type(8)))  float    v8f;
typedef __attribute__((ext_vector_type(4)))  unsigned int v4u;
typedef __attribute__((ext_vector_type(8)))  int      v8i;
typedef __attribute__((ext_vector_type(4)))  int      v4i;

#define BB    128
#define DIMC  256
#define EMBD  16
#define BOOK  1024
#define ZDIM  512
#define WD    4096
#define HD    1024  /* 2*Z */

// d_out layout (float elements), outputs concatenated in return order
#define N_WDIST  (128ull*256ull*1024ull)
#define OFF_IDX  (N_WDIST)
#define N_IDX    (128ull*256ull)
#define OFF_WREC (OFF_IDX + N_IDX)
#define N_WREC   (128ull*4096ull)
#define OFF_MU   (OFF_WREC + N_WREC)
#define N_MU     (128ull*512ull)
#define OFF_LV   (OFF_MU + N_MU)

// ---------------------------------------------------------------------------
// Kernel 1: encoder.  h = A' @ W_enc + b_enc  ->  mu / log_var.
// A'[b, e*256+d] = w_q[b, d*16+e]  (swapaxes folded into the A load).
// B (shared by all 8 waves of the block) is staged per K-chunk into LDS as
// f16, transposed [n][k] so a lane's fragment is one contiguous 32B LDS read.
// grid: 64 n-tiles, 8 waves/block = 8 m-tiles.  K = 4096 in steps of 32.
// ---------------------------------------------------------------------------
__global__ __launch_bounds__(256) void enc_kernel(const float* __restrict__ w_q,
                                                  const float* __restrict__ W_enc,
                                                  const float* __restrict__ b_enc,
                                                  float* __restrict__ out) {
    __shared__ __align__(32) _Float16 s_b[16 * 32];   // [n][k] f16, 1KB
    const int lane = threadIdx.x & 31;
    const int wave = threadIdx.x >> 5;
    const int mt = wave;               // 128/16 = 8 m-tiles
    const int nt = blockIdx.x;         // 1024/16 = 64 n-tiles
    const int lo = lane & 15;
    const int hi = lane >> 4;
    const int n0 = nt * 16;

    v8f acc = {};
    const int m = mt * 16 + lo;
    const int n = n0 + lo;
    for (int k0 = 0; k0 < WD; k0 += 32) {
        // cooperative B stage: 512 elems, 2 per thread, transposed to [n][k]
        {
            int idx = threadIdx.x;
#pragma unroll
            for (int r = 0; r < 2; ++r, idx += 256) {
                int nn = idx & 15;
                int kk = idx >> 4;
                s_b[nn * 32 + kk] = (_Float16)W_enc[(k0 + kk) * HD + n0 + nn];
            }
        }
        __syncthreads();

        v16h a;
#pragma unroll
        for (int h = 0; h < 16; ++h) {
            // A layout: lane -> M, half h -> K = 16*hi + h
            int kk = k0 + hi * 16 + h;
            int e  = kk >> 8;          // kk = e*256 + d
            int dd = kk & 255;
            a[h] = (_Float16)w_q[m * WD + dd * EMBD + e];
        }
        // B layout: lane -> N, K = 16*hi + h  -> contiguous halves in s_b
        v16h b = *(const v16h*)&s_b[lo * 32 + hi * 16];

        acc = __builtin_amdgcn_wmma_f32_16x16x32_f16(false, a, false, b,
                                                     (short)0, acc, false, false);
        __syncthreads();
    }

    // block-uniform mu / log_var select (nt<32 <=> all 16 cols < ZDIM)
    float* __restrict__ dst;
    int col;
    if (nt < 32) { dst = out + OFF_MU; col = n; }
    else         { dst = out + OFF_LV; col = n - ZDIM; }
    const float bias = b_enc[n];
#pragma unroll
    for (int v = 0; v < 8; ++v) {
        int mr = mt * 16 + v + hi * 8;       // C/D layout: VGPR v -> M
        dst[(size_t)mr * ZDIM + col] = acc[v] + bias;
    }
}

// ---------------------------------------------------------------------------
// Kernel 2: decoder.  w_recon = mu @ W_dec + b_dec.
// The 512x16 f32 column tile of W_dec (32KB) is DMA'd into LDS once per block
// by the Tensor Data Mover (one tensor_load_to_lds + s_wait_tensorcnt),
// then all 8 waves read fragments from LDS.
// grid: 256 n-tiles (4096/16), 8 waves = 8 m-tiles.  K = 512.
// ---------------------------------------------------------------------------
__global__ __launch_bounds__(256) void dec_kernel(const float* __restrict__ W_dec,
                                                  const float* __restrict__ b_dec,
                                                  float* __restrict__ out) {
    __shared__ __align__(16) float s_w[ZDIM * 16];   // 512 x 16 tile, 32KB
    const int lane = threadIdx.x & 31;
    const int wave = threadIdx.x >> 5;
    const int mt = wave;
    const int nt = blockIdx.x;
    const int lo = lane & 15;
    const int hi = lane >> 4;
    const int n0 = nt * 16;

    if (wave == 0) {
        // Tensor DMA descriptor (D#), CDNA5 ISA:
        // 2D tensor = W_dec (512 rows x 4096 cols f32), tile = 16 x 512 at col n0.
        unsigned long long ga = (unsigned long long)(size_t)(W_dec + n0);
        unsigned lds = (unsigned)(size_t)&s_w[0];
        v4u g0 = { 1u,                                   // count=1 (valid), user mode
                   lds,                                  // lds_addr
                   (unsigned)(ga & 0xFFFFFFFFull),       // global_addr[31:0]
                   (unsigned)((ga >> 32) & 0x1FFFFFFull) // global_addr[56:32]
                     | (2u << 30) };                     // type=2 ("image")
        v8i g1 = { (int)(2u << 16),          // data_size=2 -> 4 bytes
                   (int)(4096u << 16),       // tensor_dim0[15:0]=4096
                   (int)(512u << 16),        // tensor_dim0[31:16]=0 | tensor_dim1[15:0]=512
                   (int)(16u << 16),         // tensor_dim1[31:16]=0 | tile_dim0=16
                   (int)512,                 // tile_dim1=512, tile_dim2=0
                   (int)4096,                // tensor_dim0_stride[31:0]=4096
                   0, 0 };                   // stride hi / tensor_dim1_stride unused (2D)
        v4i g2 = { 0, 0, 0, 0 };             // tensor_dim2/3, dim2_stride, tile_dim3 = 0
        v4i g3 = { 0, 0, 0, 0 };
        v8i g4 = { 0, 0, 0, 0, 0, 0, 0, 0 }; // extra operand on clang-23 form (zeroed)
        __builtin_amdgcn_tensor_load_to_lds(g0, g1, g2, g3, g4, 0);
        __builtin_amdgcn_s_wait_tensorcnt((short)0);
    }
    __syncthreads();

    const float* __restrict__ mu = out + OFF_MU;
    v8f acc = {};
    const int m = mt * 16 + lo;
    const int n = n0 + lo;
    for (int k0 = 0; k0 < ZDIM; k0 += 32) {
        v16h a, b;
#pragma unroll
        for (int h = 0; h < 16; ++h) {
            int kk = k0 + hi * 16 + h;
            a[h] = (_Float16)mu[(size_t)m * ZDIM + kk];
        }
#pragma unroll
        for (int h = 0; h < 16; ++h) {
            int kk = k0 + hi * 16 + h;       // tile is row-major [k][n] in LDS
            b[h] = (_Float16)s_w[kk * 16 + lo];
        }
        acc = __builtin_amdgcn_wmma_f32_16x16x32_f16(false, a, false, b,
                                                     (short)0, acc, false, false);
    }
    const float bias = b_dec[n];
#pragma unroll
    for (int v = 0; v < 8; ++v) {
        int mr = mt * 16 + v + hi * 8;
        out[OFF_WREC + (size_t)mr * WD + n] = acc[v] + bias;
    }
}

// ---------------------------------------------------------------------------
// Kernel 3: fused distance + argmin.
// For each (d, m-tile): cross = X_d(16x16) @ C_d^T(16x1024) via WMMA (K padded
// to 32), dist = x2 - 2*cross + c2 written exactly once (the 134MB output is
// the roofline bottleneck), argmin via packed u64 LDS ds_min_u64.
// grid: (256 d, 8 m-tiles), 8 waves each handling 8 n-tiles of 16.
// ---------------------------------------------------------------------------
__global__ __launch_bounds__(256) void dist_kernel(const float* __restrict__ codebook,
                                                   float* __restrict__ out) {
    __shared__ float s_c2[BOOK];
    __shared__ float s_x2[16];
    __shared__ unsigned long long s_min[16];

    const int d    = blockIdx.x;
    const int mt   = blockIdx.y;
    const int tid  = threadIdx.x;
    const int lane = tid & 31;
    const int wave = tid >> 5;
    const int lo   = lane & 15;
    const int hi   = lane >> 4;

    const float* __restrict__ wrec = out + OFF_WREC;
    const float* __restrict__ cb   = codebook + (size_t)d * BOOK * EMBD;

    for (int k = tid; k < BOOK; k += 256) {
        float s = 0.f;
#pragma unroll
        for (int e = 0; e < EMBD; ++e) { float c = cb[k * EMBD + e]; s += c * c; }
        s_c2[k] = s;
    }
    if (tid < 16) {
        int m = mt * 16 + tid;
        float s = 0.f;
#pragma unroll
        for (int e = 0; e < EMBD; ++e) {
            float x = wrec[(size_t)m * WD + d * EMBD + e];
            s += x * x;
        }
        s_x2[tid] = s;
        s_min[tid] = ~0ull;
    }
    __syncthreads();

    // A fragment reused across all n-tiles: rows of X_d, K=16 padded to 32.
    v16h a;
#pragma unroll
    for (int h = 0; h < 16; ++h) {
        int kk = hi * 16 + h;
        a[h] = (kk < EMBD)
                 ? (_Float16)wrec[(size_t)(mt * 16 + lo) * WD + d * EMBD + kk]
                 : (_Float16)0.f;
    }

    for (int t = 0; t < 8; ++t) {
        const int nt = wave * 8 + t;
        v16h b;
#pragma unroll
        for (int h = 0; h < 16; ++h) {
            int kk = hi * 16 + h;            // lanes 16..31 carry K>=16 -> zero pad
            b[h] = (kk < EMBD) ? (_Float16)cb[(nt * 16 + lo) * EMBD + kk]
                               : (_Float16)0.f;
        }
        v8f c = {};
        c = __builtin_amdgcn_wmma_f32_16x16x32_f16(false, a, false, b,
                                                   (short)0, c, false, false);
        const int n = nt * 16 + lo;
        const float c2v = s_c2[n];
#pragma unroll
        for (int v = 0; v < 8; ++v) {
            int ml = v + hi * 8;
            float dval = s_x2[ml] - 2.0f * c[v] + c2v;
            int m = mt * 16 + ml;
            out[(size_t)m * (DIMC * BOOK) + (size_t)d * BOOK + n] = dval;
            unsigned long long key =
                ((unsigned long long)__float_as_uint(dval) << 32) | (unsigned)n;
            atomicMin(&s_min[ml], key);      // ds_min_u64; low k wins ties
        }
    }
    __syncthreads();
    if (tid < 16) {
        int m = mt * 16 + tid;
        unsigned k = (unsigned)(s_min[tid] & 0xFFFFFFFFu);
        out[OFF_IDX + (size_t)m * DIMC + d] = (float)k;
    }
}

extern "C" void kernel_launch(void* const* d_in, const int* in_sizes, int n_in,
                              void* d_out, int out_size, void* d_ws, size_t ws_size,
                              hipStream_t stream) {
    (void)in_sizes; (void)n_in; (void)out_size; (void)d_ws; (void)ws_size;
    const float* w_q      = (const float*)d_in[0];
    // d_in[1] = pseudo_inputs: provably unused (reference drops h[:-NPSEUDO]).
    const float* W_enc    = (const float*)d_in[2];
    const float* b_enc    = (const float*)d_in[3];
    const float* W_dec    = (const float*)d_in[4];
    const float* b_dec    = (const float*)d_in[5];
    const float* codebook = (const float*)d_in[6];
    float* out = (float*)d_out;

    enc_kernel<<<64, 256, 0, stream>>>(w_q, W_enc, b_enc, out);
    dec_kernel<<<256, 256, 0, stream>>>(W_dec, b_dec, out);
    dist_kernel<<<dim3(256, 8), 256, 0, stream>>>(codebook, out);
}
